// ProjectiveGraphSAGE_66383014527509
// MI455X (gfx1250) — compile-verified, matrix operand order checked by reading, not run on hardware
//
#include <hip/hip_runtime.h>

typedef _Float16 v16h __attribute__((ext_vector_type(16)));
typedef _Float16 v2h  __attribute__((ext_vector_type(2)));
typedef float    v8f  __attribute__((ext_vector_type(8)));

#define M_NODES 100000
#define IN_CH   129
#define C       128

// ---------------- utility kernels ----------------

__global__ void zero_f32(float* __restrict__ p, int n) {
  int i = blockIdx.x * blockDim.x + threadIdx.x;
  int stride = gridDim.x * blockDim.x;
  for (; i < n; i += stride) p[i] = 0.0f;
}

// strip the pass-through homogeneous column: x (M x 129) -> xs (M x 128), aligned
__global__ void pack_spatial(const float* __restrict__ x, float* __restrict__ xs, int M) {
  int t = blockIdx.x * blockDim.x + threadIdx.x;   // one thread per 4 channels
  if (t >= M * (C / 4)) return;
  int node = t >> 5;           // C/4 == 32
  int c4 = (t & 31) * 4;
  const float* s = x + (size_t)node * IN_CH + c4;  // stride-129 source (only 4B aligned)
  float4 v = make_float4(s[0], s[1], s[2], s[3]);
  *(float4*)(xs + (size_t)node * C + c4) = v;
}

__global__ void count_deg(const int* __restrict__ dst, unsigned* __restrict__ cnt, int E) {
  int e = blockIdx.x * blockDim.x + threadIdx.x;
  if (e < E) atomicAdd(&cnt[dst[e]], 1u);
}

__global__ void make_inv(const unsigned* __restrict__ cnt, float* __restrict__ inv, int M) {
  int i = blockIdx.x * blockDim.x + threadIdx.x;
  if (i < M) {
    unsigned c = cnt[i];
    inv[i] = 1.0f / (float)(c > 1u ? c : 1u);
  }
}

// one wave per edge; 32 lanes x float4 = 128 channels; L2-resident f32 atomics
__global__ void scatter_add(const float* __restrict__ feat, const int* __restrict__ src,
                            const int* __restrict__ dst, float* __restrict__ agg, int E) {
  int wave = (int)((blockIdx.x * blockDim.x + threadIdx.x) >> 5);
  int lane = threadIdx.x & 31;
  if (wave >= E) return;
  int s = src[wave];
  int d = dst[wave];
  float4 v = *(const float4*)(feat + (size_t)s * C + lane * 4);
  float* o = agg + (size_t)d * C + lane * 4;
  unsafeAtomicAdd(o + 0, v.x);
  unsafeAtomicAdd(o + 1, v.y);
  unsafeAtomicAdd(o + 2, v.z);
  unsafeAtomicAdd(o + 3, v.w);
}

// ---------------- WMMA dual-GEMM ----------------
// out = As @ Ws + (Ag * inv) @ Wn + bias, optional ReLU.
// Block: 8 waves; wave w owns N-tile w (16 cols); block owns a 16-row M-tile.
//
// Weights are staged into LDS pre-swizzled into the exact 16-bit 16x32 B-fragment
// layout (ISA 7.12.2): lane L holds column N = L&15 with K-pairs
// {0..7,16..23} (L<16) / {8..15,24..31} (L>=16). LDS slot for half j of lane L,
// k-step ks, n-tile nt:  (((nt*4 + ks)*32 + L)*16 + j). Each lane then reads its
// whole fragment as one contiguous 32B chunk (2x ds_load_b128).

__device__ __forceinline__ int kmap(int half, int j) {
  int p = j >> 1;
  int base = (p < 4) ? (half * 8 + 2 * p) : (16 + half * 8 + 2 * (p - 4));
  return base + (j & 1);
}

__global__ __launch_bounds__(256)
void sage_gemm(const float* __restrict__ As, const float* __restrict__ Ag,
               const float* __restrict__ inv, const float* __restrict__ Ws,
               const float* __restrict__ Wn, const float* __restrict__ bias,
               float* __restrict__ out, int relu) {
  __shared__ _Float16 lw[C * C];            // 32 KB swizzled weight fragments

  int wave = threadIdx.x >> 5;
  int lane = threadIdx.x & 31;
  int half = lane >> 4;
  int nIdx = lane & 15;
  int mBase = blockIdx.x * 16;
  int row  = mBase + nIdx;                  // A-fragment row for this lane
  int nCol = wave * 16 + nIdx;              // B/C column for this lane

  // A-side K index map (pairs are contiguous -> b128-mergeable global loads)
  int kk[8];
#pragma unroll
  for (int p = 0; p < 8; ++p)
    kk[p] = (p < 4) ? (half * 8 + 2 * p) : (16 + half * 8 + 2 * (p - 4));

  const v16h* frag = (const v16h*)lw;       // 32B fragment per (ntile,kstep,lane)

  v8f acc = {};

  // ================= phase 1: spatial @ Ws =================
  for (int t = threadIdx.x; t < (C * C) / 2; t += 256) {   // one u32 slot (2 halves)
    int d      = t * 2;
    int j      = d & 15;
    int lane_s = (d >> 4) & 31;
    int kstep  = (d >> 9) & 3;
    int ntile  = d >> 11;
    int k      = kstep * 32 + kmap(lane_s >> 4, j);
    int n      = ntile * 16 + (lane_s & 15);
    v2h hv = { (_Float16)Ws[k * C + n], (_Float16)Ws[(k + 1) * C + n] };
    *(v2h*)(&lw[d]) = hv;
  }
  __syncthreads();
  {
    const float* Ar = As + (size_t)row * C;
#pragma unroll
    for (int ks = 0; ks < 4; ++ks) {
      v16h a;
#pragma unroll
      for (int p = 0; p < 8; ++p) {
        float2 t = *(const float2*)(Ar + ks * 32 + kk[p]);
        a[2 * p]     = (_Float16)t.x;
        a[2 * p + 1] = (_Float16)t.y;
      }
      v16h b = frag[(wave * 4 + ks) * 32 + lane];
      acc = __builtin_amdgcn_wmma_f32_16x16x32_f16(false, a, false, b,
                                                   (short)0, acc, false, false);
    }
  }
  __syncthreads();

  // ================= phase 2: (agg * inv) @ Wn =================
  for (int t = threadIdx.x; t < (C * C) / 2; t += 256) {
    int d      = t * 2;
    int j      = d & 15;
    int lane_s = (d >> 4) & 31;
    int kstep  = (d >> 9) & 3;
    int ntile  = d >> 11;
    int k      = kstep * 32 + kmap(lane_s >> 4, j);
    int n      = ntile * 16 + (lane_s & 15);
    v2h hv = { (_Float16)Wn[k * C + n], (_Float16)Wn[(k + 1) * C + n] };
    *(v2h*)(&lw[d]) = hv;
  }
  __syncthreads();
  {
    float ir = inv[row];
    const float* Gr = Ag + (size_t)row * C;
#pragma unroll
    for (int ks = 0; ks < 4; ++ks) {
      v16h a;
#pragma unroll
      for (int p = 0; p < 8; ++p) {
        float2 t = *(const float2*)(Gr + ks * 32 + kk[p]);
        a[2 * p]     = (_Float16)(t.x * ir);
        a[2 * p + 1] = (_Float16)(t.y * ir);
      }
      v16h b = frag[(wave * 4 + ks) * 32 + lane];
      acc = __builtin_amdgcn_wmma_f32_16x16x32_f16(false, a, false, b,
                                                   (short)0, acc, false, false);
    }
  }

  // epilogue: C/D layout -> VGPR r holds M = mBase + 8*half + r, N = nCol
  float bv = bias[nCol];
#pragma unroll
  for (int r = 0; r < 8; ++r) {
    float v = acc[r] + bv;
    if (relu) v = fmaxf(v, 0.0f);
    out[(size_t)(mBase + half * 8 + r) * C + nCol] = v;
  }
}

// ---------------- host launch ----------------

extern "C" void kernel_launch(void* const* d_in, const int* in_sizes, int n_in,
                              void* d_out, int out_size, void* d_ws, size_t ws_size,
                              hipStream_t stream) {
  const float* x   = (const float*)d_in[0];
  const int*   ei  = (const int*)d_in[1];
  const float* w1s = (const float*)d_in[2];
  const float* w1n = (const float*)d_in[3];
  const float* b1  = (const float*)d_in[4];
  const float* w2s = (const float*)d_in[5];
  const float* w2n = (const float*)d_in[6];
  const float* b2  = (const float*)d_in[7];
  float* out = (float*)d_out;

  const int M = M_NODES;
  const int E = in_sizes[1] / 2;
  const int* src = ei;
  const int* dst = ei + E;

  // workspace layout
  float*    xs  = (float*)d_ws;              // M*128
  float*    h1  = xs + (size_t)M * C;        // M*128
  float*    agg = h1 + (size_t)M * C;        // M*128
  unsigned* cnt = (unsigned*)(agg + (size_t)M * C);  // M
  float*    inv = (float*)(cnt + M);         // M

  // pack spatial features (drop pass-through homog column)
  pack_spatial<<<(M * 32 + 255) / 256, 256, 0, stream>>>(x, xs, M);

  // degrees + reciprocal (graph is fixed -> once per call)
  zero_f32<<<256, 256, 0, stream>>>((float*)cnt, M);
  count_deg<<<(E + 255) / 256, 256, 0, stream>>>(dst, cnt, E);
  make_inv<<<(M + 255) / 256, 256, 0, stream>>>(cnt, inv, M);

  // layer 1
  zero_f32<<<2048, 256, 0, stream>>>(agg, M * C);
  scatter_add<<<(E + 7) / 8, 256, 0, stream>>>(xs, src, dst, agg, E);
  sage_gemm<<<M / 16, 256, 0, stream>>>(xs, agg, inv, w1s, w1n, b1, h1, 1);

  // layer 2 (writes final 100000x128 f32 directly to d_out)
  zero_f32<<<2048, 256, 0, stream>>>(agg, M * C);
  scatter_add<<<(E + 7) / 8, 256, 0, stream>>>(h1, src, dst, agg, E);
  sage_gemm<<<M / 16, 256, 0, stream>>>(h1, agg, inv, w2s, w2n, b2, out, 0);
}
